// BitwiseMultipyLogis_25649544691947
// MI455X (gfx1250) — matrix operand order repaired and checked by reading, not compiled, not run on hardware
//
#include <hip/hip_runtime.h>

// CDNA5 (gfx1250) fused kernel for BitwiseMultipyLogis.
// One HBM pass over node_features; bf16 WMMA (v_wmma_f32_16x16x32_bf16) for
// both GEMMs with f32 accumulation. trans/theta are pre-converted to bf16 and
// pre-swizzled into the per-lane WMMA B layout (module __device__ arrays,
// regenerated every call) so each B chunk is one contiguous 32-byte per-lane
// load. f32->bf16 packing uses v_fma_mixlo/mixhi_bf16 (1 VALU op per element,
// fused multiply for the bitwise phase); tanh uses the native v_tanh_f32
// TRANS op. proj is shared across the 8 layer-waves via LDS.

typedef __attribute__((ext_vector_type(16))) __bf16 v16bf;
typedef __attribute__((ext_vector_type(8)))  float  v8f;

#define NLAYERS 8
#define FDIM    128
#define TILE_M  16
#define PROJ_PITCH 132      // pad f32 rows: avoid 16-way LDS bank conflicts
#define MAT_ELEMS (FDIM * FDIM)   // 16384 elems per 128x128 matrix

// Swizzled bf16 B operands, regenerated from the inputs on every launch.
__device__ __attribute__((aligned(256))) __bf16 g_transB[MAT_ELEMS];
__device__ __attribute__((aligned(256))) __bf16 g_thetaB[NLAYERS * MAT_ELEMS];

__device__ __forceinline__ __bf16 cvt_bf16(float f) {
    // round-to-nearest-even f32 -> bf16 (used only in the tiny swizzle kernel)
    unsigned u = __builtin_bit_cast(unsigned, f);
    u += 0x7FFFu + ((u >> 16) & 1u);
    unsigned short h = (unsigned short)(u >> 16);
    return __builtin_bit_cast(__bf16, h);
}

// Pack {bf16(f0), bf16(f1)} into one dword: v_fma_mixlo/mixhi_bf16 (VOP3P).
__device__ __forceinline__ unsigned pack2_bf16(float f0, float f1) {
    unsigned d;
    asm("v_fma_mixlo_bf16 %0, %1, 1.0, 0" : "=v"(d) : "v"(f0));
    asm("v_fma_mixhi_bf16 %0, %1, 1.0, 0" : "+v"(d) : "v"(f1));
    return d;
}

// Pack {bf16(a0*b0), bf16(a1*b1)} into one dword (multiply fused into cvt).
__device__ __forceinline__ unsigned mul_pack2_bf16(float a0, float b0,
                                                   float a1, float b1) {
    unsigned d;
    asm("v_fma_mixlo_bf16 %0, %1, %2, 0" : "=v"(d) : "v"(a0), "v"(b0));
    asm("v_fma_mixhi_bf16 %0, %1, %2, 0" : "+v"(d) : "v"(a1), "v"(b1));
    return d;
}

__device__ __forceinline__ float hw_tanh(float v) {
    float r;
    asm("v_tanh_f32 %0, %1" : "=v"(r) : "v"(v));   // native CDNA5 TRANS op
    return r;
}

union bfpack16 { unsigned u[8]; v16bf v; };

// Swizzled B layout: for a 128x128 matrix M, chunk (j, g) and lane hold 16
// contiguous bf16 at offset ((j*4+g)*32 + lane)*16; element e maps to M[k][c]
// with c = 16*j + (lane&15), k = 32*g + 8*(lane>>4) + (e&7) + 16*(e>>3).
__global__ __launch_bounds__(256)
void bml_swizzle_bf16_kernel(const float* __restrict__ trans,  // [F,F]
                             const float* __restrict__ theta,  // [L,F,F]
                             __bf16* __restrict__ transB,
                             __bf16* __restrict__ thetaB)
{
    const int tid = blockIdx.x * blockDim.x + threadIdx.x;
    if (tid >= (NLAYERS + 1) * MAT_ELEMS) return;
    const int mat  = tid / MAT_ELEMS;    // 0..7 = theta layer, 8 = trans
    const int rem  = tid % MAT_ELEMS;
    const int e    = rem & 15;
    const int lane = (rem >> 4) & 31;
    const int g    = (rem >> 9) & 3;
    const int j    = rem >> 11;
    const int c = 16 * j + (lane & 15);
    const int k = 32 * g + 8 * (lane >> 4) + (e & 7) + 16 * (e >> 3);
    if (mat == NLAYERS) {
        transB[rem] = cvt_bf16(trans[k * FDIM + c]);
    } else {
        thetaB[mat * MAT_ELEMS + rem] =
            cvt_bf16(theta[mat * MAT_ELEMS + k * FDIM + c]);
    }
}

__global__ __launch_bounds__(256)
void bml_fused_kernel(const float* __restrict__ x,        // [L,N,F]
                      const float* __restrict__ bias,     // [F]
                      const float* __restrict__ logw,     // [F]  (logis_w[0])
                      const float* __restrict__ logb,     // [1]
                      const int*   __restrict__ lp_ptr,   // scalar layer_predict
                      const __bf16* __restrict__ transB,  // swizzled bf16
                      const __bf16* __restrict__ thetaB,  // swizzled bf16
                      float* __restrict__ out,            // [N,F]
                      int N)
{
    __shared__ float s_proj[NLAYERS][TILE_M][PROJ_PITCH];
    __shared__ float s_scores[NLAYERS][TILE_M];

    const int tid  = threadIdx.x;
    const int l    = tid >> 5;        // wave id == layer
    const int lane = tid & 31;
    const int hlf  = lane >> 4;
    const int n16  = lane & 15;
    const int n0   = blockIdx.x * TILE_M;

    int lp = lp_ptr[0];
    lp = (lp < 0) ? 0 : (lp >= NLAYERS ? NLAYERS - 1 : lp);

    // ---------------- Phase 1: proj_l = tanh(X_l @ trans + bias) -------------
    // A (16x128 bf16) in WMMA A layout: lane m=lane&15;
    // e=0..7 -> K = 32g + 8*hlf + e ; e=8..15 -> K += 16.
    v16bf a[4];
    {
        int row = n0 + n16;
        if (row >= N) row = N - 1;
        const float4* x4 = reinterpret_cast<const float4*>(
                               x + ((size_t)l * N + row) * FDIM);
        #pragma unroll
        for (int g = 0; g < 4; ++g) {
            const int q = (32 * g + (hlf ? 8 : 0)) >> 2;   // float4 index
            float4 A0 = x4[q],     A1 = x4[q + 1];
            float4 B0 = x4[q + 4], B1 = x4[q + 5];
            bfpack16 pa;
            pa.u[0] = pack2_bf16(A0.x, A0.y); pa.u[1] = pack2_bf16(A0.z, A0.w);
            pa.u[2] = pack2_bf16(A1.x, A1.y); pa.u[3] = pack2_bf16(A1.z, A1.w);
            pa.u[4] = pack2_bf16(B0.x, B0.y); pa.u[5] = pack2_bf16(B0.z, B0.w);
            pa.u[6] = pack2_bf16(B1.x, B1.y); pa.u[7] = pack2_bf16(B1.z, B1.w);
            a[g] = pa.v;
        }
    }

    #pragma unroll
    for (int j = 0; j < 8; ++j) {                     // output col tile
        v8f acc = {};
        #pragma unroll
        for (int g = 0; g < 4; ++g) {
            const v16bf b = *reinterpret_cast<const v16bf*>(
                                transB + (((j * 4 + g) * 32) + lane) * 16);
            acc = __builtin_amdgcn_wmma_f32_16x16x32_bf16(false, a[g], false, b,
                                                          (short)0, acc, false, false);
        }
        const float bv = bias[16 * j + n16];
        #pragma unroll
        for (int r = 0; r < 8; ++r) {
            const float p = hw_tanh(acc[r] + bv);
            s_proj[l][r + 8 * hlf][16 * j + n16] = p;   // M = r+8*hlf, col = 16j+n16
        }
    }
    __syncthreads();

    // ------- Phase 2: bitwise = proj_l * proj_lp, repacked to bf16 A layout --
    // v_fma_mixlo/hi_bf16 fuses the product with the convert+pack.
    {
        const float4* pl4  = reinterpret_cast<const float4*>(&s_proj[l][n16][0]);
        const float4* plp4 = reinterpret_cast<const float4*>(&s_proj[lp][n16][0]);
        #pragma unroll
        for (int g = 0; g < 4; ++g) {
            const int q = (32 * g + (hlf ? 8 : 0)) >> 2;
            float4 a0 = pl4[q],      a1 = pl4[q + 1];
            float4 b0 = pl4[q + 4],  b1 = pl4[q + 5];
            float4 c0 = plp4[q],     c1 = plp4[q + 1];
            float4 d0 = plp4[q + 4], d1 = plp4[q + 5];
            bfpack16 pa;
            pa.u[0] = mul_pack2_bf16(a0.x, c0.x, a0.y, c0.y);
            pa.u[1] = mul_pack2_bf16(a0.z, c0.z, a0.w, c0.w);
            pa.u[2] = mul_pack2_bf16(a1.x, c1.x, a1.y, c1.y);
            pa.u[3] = mul_pack2_bf16(a1.z, c1.z, a1.w, c1.w);
            pa.u[4] = mul_pack2_bf16(b0.x, d0.x, b0.y, d0.y);
            pa.u[5] = mul_pack2_bf16(b0.z, d0.z, b0.w, d0.w);
            pa.u[6] = mul_pack2_bf16(b1.x, d1.x, b1.y, d1.y);
            pa.u[7] = mul_pack2_bf16(b1.z, d1.z, b1.w, d1.w);
            a[g] = pa.v;
        }
    }

    // ------- Phase 3: y = bitwise @ theta_l, fold logistic dot in epilogue ---
    float spart[8];
    #pragma unroll
    for (int r = 0; r < 8; ++r) spart[r] = 0.0f;
    const __bf16* thB = thetaB + (size_t)l * MAT_ELEMS;

    #pragma unroll
    for (int j = 0; j < 8; ++j) {
        v8f acc = {};
        #pragma unroll
        for (int g = 0; g < 4; ++g) {
            const v16bf b = *reinterpret_cast<const v16bf*>(
                                thB + (((j * 4 + g) * 32) + lane) * 16);
            acc = __builtin_amdgcn_wmma_f32_16x16x32_bf16(false, a[g], false, b,
                                                          (short)0, acc, false, false);
        }
        const float wl = logw[16 * j + n16];
        #pragma unroll
        for (int r = 0; r < 8; ++r)
            spart[r] += acc[r] * wl;
    }

    // butterfly-reduce each spart[r] within the 16-lane half (wave32 native)
    #pragma unroll
    for (int r = 0; r < 8; ++r) {
        float v = spart[r];
        v += __shfl_xor(v, 1, 16);
        v += __shfl_xor(v, 2, 16);
        v += __shfl_xor(v, 4, 16);
        v += __shfl_xor(v, 8, 16);
        spart[r] = v;
    }
    const float lb = logb[0];
    if ((lane & 15) == 0) {                // lane 0 -> m=r, lane 16 -> m=r+8
        #pragma unroll
        for (int r = 0; r < 8; ++r) {
            const float s = spart[r] + lb;
            s_scores[l][r + 8 * hlf] = __builtin_amdgcn_rcpf(1.0f + __expf(-s));
        }
    }
    __syncthreads();

    // ------- Phase 4: softmax over layers + weighted aggregation -------------
    {
        const int m  = tid >> 4;           // node within tile (0..15)
        const int f0 = (tid & 15) * 8;     // 8 contiguous floats per thread
        float sc[NLAYERS];
        float mx = -1e30f;
        #pragma unroll
        for (int q = 0; q < NLAYERS; ++q) { sc[q] = s_scores[q][m]; mx = fmaxf(mx, sc[q]); }
        float ssum = 0.0f;
        #pragma unroll
        for (int q = 0; q < NLAYERS; ++q) { sc[q] = __expf(sc[q] - mx); ssum += sc[q]; }
        const float inv = __builtin_amdgcn_rcpf(ssum);
        #pragma unroll
        for (int q = 0; q < NLAYERS; ++q) sc[q] *= inv;

        if (n0 + m < N) {
            const float4* base = reinterpret_cast<const float4*>(&s_proj[lp][m][f0]);
            float4 r0 = base[0], r1 = base[1];
            #pragma unroll
            for (int q = 0; q < NLAYERS; ++q) {
                const float4* p = reinterpret_cast<const float4*>(&s_proj[q][m][f0]);
                float4 p0 = p[0], p1 = p[1];
                const float w = sc[q];
                r0.x += w * p0.x; r0.y += w * p0.y; r0.z += w * p0.z; r0.w += w * p0.w;
                r1.x += w * p1.x; r1.y += w * p1.y; r1.z += w * p1.z; r1.w += w * p1.w;
            }
            float4* orow = reinterpret_cast<float4*>(out + (size_t)(n0 + m) * FDIM + f0);
            orow[0] = r0;
            orow[1] = r1;
        }
    }
}

extern "C" void kernel_launch(void* const* d_in, const int* in_sizes, int n_in,
                              void* d_out, int out_size, void* d_ws, size_t ws_size,
                              hipStream_t stream) {
    const float* x     = (const float*)d_in[0];  // node_features [L,N,F]
    const float* trans = (const float*)d_in[1];  // [F,F]
    const float* bias  = (const float*)d_in[2];  // [F]
    const float* theta = (const float*)d_in[3];  // [L,F,F]
    const float* logw  = (const float*)d_in[4];  // [1,F]
    const float* logb  = (const float*)d_in[5];  // [1]
    const int*   lp    = (const int*)d_in[6];    // scalar
    float* out = (float*)d_out;

    // Resolve device-global addresses once; pass as kernargs so B loads are
    // saddr+imm (no per-load s_get_pc/s_add relocation SALU in the hot loop).
    void* transB_dev = nullptr;
    void* thetaB_dev = nullptr;
    hipGetSymbolAddress(&transB_dev, HIP_SYMBOL(g_transB));
    hipGetSymbolAddress(&thetaB_dev, HIP_SYMBOL(g_thetaB));

    const int N = in_sizes[0] / (NLAYERS * FDIM);
    const int blocks = (N + TILE_M - 1) / TILE_M;

    const int conv_elems  = (NLAYERS + 1) * MAT_ELEMS;
    const int conv_blocks = (conv_elems + 255) / 256;
    bml_swizzle_bf16_kernel<<<conv_blocks, 256, 0, stream>>>(
        trans, theta, (__bf16*)transB_dev, (__bf16*)thetaB_dev);
    bml_fused_kernel<<<blocks, 256, 0, stream>>>(
        x, bias, logw, logb, lp,
        (const __bf16*)transB_dev, (const __bf16*)thetaB_dev, out, N);
}
